// AutoregressiveTransformer_81870666596920
// MI455X (gfx1250) — matrix operand order repaired and checked by reading, not compiled, not run on hardware
//
#include <hip/hip_runtime.h>
#include <hip/hip_bf16.h>
#include <math.h>

// ---------------------------------------------------------------------------
// Model constants (match reference)
// ---------------------------------------------------------------------------
#define DMODEL 512
#define NHEAD  8
#define DHEAD  64
#define LLANE  12
#define BATCH  64
#define LATENT 64

typedef __attribute__((ext_vector_type(16))) __bf16 v16bf;
typedef __attribute__((ext_vector_type(8)))  float  v8f;
typedef __attribute__((ext_vector_type(4)))  float  v4f;

// ---------------------------------------------------------------------------
// GEMM:  C[M,N] = act(A[M,K] @ W[K,N] + bias[N])
//
// Constraints guaranteed by the caller (true for every GEMM in this model):
//   M % 64 == 0, K % 32 == 0. N is arbitrary (store-guarded / load-clamped).
//
// Block = 128 threads = 4 waves. Block tile = 64(M) x 64(N).
//   - A slab [64 x 32] f32 staged in LDS via coalesced b128 loads (A is read
//     once per 64-row stripe -> flatten W1 stream stays single-pass).
//   - Each wave owns a 16-column slice, keeps 4 M-tile f32 accumulators and
//     one bf16 B fragment per K-step -> 4x v_wmma_f32_16x16x32_bf16 per step.
//   - No divergent branches in the K loop: EXEC is all-ones at every WMMA.
// act: 0 = none, 1 = relu, 2 = exact gelu
// ---------------------------------------------------------------------------
#define LDS_STRIDE 36   // 36 floats = 144B: 16B-aligned rows, staggers banks

__global__ void gemm_wmma_kernel(const float* __restrict__ A,
                                 const float* __restrict__ W,
                                 const float* __restrict__ bias,
                                 float* __restrict__ C,
                                 int M, int N, int K, int act)
{
    __shared__ float As[64 * LDS_STRIDE];

    const int wave = threadIdx.x >> 5;
    const int lane = threadIdx.x & 31;
    const int n0    = blockIdx.x * 64 + wave * 16;   // this wave's column tile
    const int mrow0 = blockIdx.y * 64;               // this block's row stripe
    const bool active = (n0 < N);

    const int g  = lane >> 4;                        // lane group (0/1)
    const int n  = n0 + (lane & 15);                 // C column for this lane
    const int nl = (n < N) ? n : (N - 1);            // clamped load column

    // Cooperative A-slab load mapping: thread -> (row, 16-float half-row)
    const int ar = threadIdx.x >> 1;
    const int ac = (threadIdx.x & 1) << 4;
    const float* Asrc = A + (size_t)(mrow0 + ar) * K + ac;
    float*       Adst = &As[ar * LDS_STRIDE + ac];

    // Per-wave A-fragment LDS base: row (lane&15) within each 16-row tile,
    // K runs {g*8..g*8+7} and {16+g*8..16+g*8+7} (ISA 7.12.2 bf16 A layout).
    const float* Afrag = &As[(lane & 15) * LDS_STRIDE + (g << 3)];

    const float* Wp = W + nl;                        // column pointer
    v8f acc[4] = {{}, {}, {}, {}};

    for (int k0 = 0; k0 < K; k0 += 32) {
        // ---- stage A[mrow0..+64][k0..+32] into LDS (coalesced b128) ----
        const v4f* src = (const v4f*)(Asrc + k0);
        v4f*       dst = (v4f*)Adst;
        dst[0] = src[0]; dst[1] = src[1]; dst[2] = src[2]; dst[3] = src[3];
        __syncthreads();

        if (active) {
            // ---- B fragment: 16 coalesced b32 loads, cvt to bf16 ----
            v16bf b;
            const size_t krow = (size_t)(k0 + (g << 3)) * N;
#pragma unroll
            for (int s = 0; s < 16; ++s)
                b[s] = (__bf16)Wp[krow + (size_t)(s + (s & 8)) * N];
            if (k0 + 32 < K)
                __builtin_prefetch(&Wp[(size_t)(k0 + 32 + (g << 4)) * N], 0, 1);

            // ---- 4 M-tiles: A fragments from LDS, accumulate ----
#pragma unroll
            for (int mt = 0; mt < 4; ++mt) {
                const float* af = Afrag + mt * 16 * LDS_STRIDE;
                v4f la0 = *(const v4f*)(af);
                v4f la1 = *(const v4f*)(af + 4);
                v4f la2 = *(const v4f*)(af + 16);
                v4f la3 = *(const v4f*)(af + 20);
                v16bf a;
                a[0]  = (__bf16)la0.x; a[1]  = (__bf16)la0.y;
                a[2]  = (__bf16)la0.z; a[3]  = (__bf16)la0.w;
                a[4]  = (__bf16)la1.x; a[5]  = (__bf16)la1.y;
                a[6]  = (__bf16)la1.z; a[7]  = (__bf16)la1.w;
                a[8]  = (__bf16)la2.x; a[9]  = (__bf16)la2.y;
                a[10] = (__bf16)la2.z; a[11] = (__bf16)la2.w;
                a[12] = (__bf16)la3.x; a[13] = (__bf16)la3.y;
                a[14] = (__bf16)la3.z; a[15] = (__bf16)la3.w;
                acc[mt] = __builtin_amdgcn_wmma_f32_16x16x32_bf16(
                              false, a, false, b, (short)0, acc[mt],
                              false, false);
            }
        }
        __syncthreads();
    }

    if (!active) return;

    // Epilogue: C/D layout — VGPR j -> row (tile base + j + 8*g), col n.
    const float bb = bias ? bias[nl] : 0.0f;
    if (n < N) {
#pragma unroll
        for (int mt = 0; mt < 4; ++mt) {
            const int rbase = mrow0 + mt * 16 + (g << 3);
#pragma unroll
            for (int j = 0; j < 8; ++j) {
                float v = acc[mt][j] + bb;
                if (act == 1)      v = fmaxf(v, 0.0f);
                else if (act == 2) v = 0.5f * v * (1.0f + erff(v * 0.70710678118654752f));
                C[(size_t)(rbase + j) * N + n] = v;
            }
        }
    }
}

// ---------------------------------------------------------------------------
// Small multi-head attention: O = softmax(QK^T / 8) V.
// Token rows laid out [t*BATCH + b, DMODEL]. One block = one (si, b, h);
// 64 threads, thread d handles head dimension d. kvlen <= 12.
// ---------------------------------------------------------------------------
__global__ void attn_kernel(const float* __restrict__ Q,
                            const float* __restrict__ K,
                            const float* __restrict__ V,
                            float* __restrict__ O,
                            int qlen, int kvlen)
{
    const int h  = blockIdx.x % NHEAD;
    const int b  = (blockIdx.x / NHEAD) % BATCH;
    const int si = blockIdx.x / (NHEAD * BATCH);
    const int d  = threadIdx.x;

    __shared__ float red[64];
    __shared__ float sc[16];

    const size_t qoff = (size_t)(si * BATCH + b) * DMODEL + h * DHEAD;
    const float  q    = Q[qoff + d];

    for (int t = 0; t < kvlen; ++t) {
        red[d] = q * K[(size_t)(t * BATCH + b) * DMODEL + h * DHEAD + d];
        __syncthreads();
        for (int s = 32; s > 0; s >>= 1) {
            if (d < s) red[d] += red[d + s];
            __syncthreads();
        }
        if (d == 0) sc[t] = red[0] * 0.125f;   // / sqrt(64)
        __syncthreads();
    }

    float mx = -3.4e38f;
    for (int t = 0; t < kvlen; ++t) mx = fmaxf(mx, sc[t]);
    float sum = 0.0f;
    for (int t = 0; t < kvlen; ++t) sum += __expf(sc[t] - mx);
    float o = 0.0f;
    for (int t = 0; t < kvlen; ++t)
        o += __expf(sc[t] - mx) * V[(size_t)(t * BATCH + b) * DMODEL + h * DHEAD + d];
    O[qoff + d] = o / sum;
}

// ---------------------------------------------------------------------------
// out = LayerNorm(x + r) * g + b over DMODEL=512; r may be nullptr.
// One block (256 threads) per row; safe in place.
// ---------------------------------------------------------------------------
__global__ void add_ln_kernel(float* __restrict__ out,
                              const float* __restrict__ x,
                              const float* __restrict__ r,
                              const float* __restrict__ g,
                              const float* __restrict__ b,
                              int rows)
{
    const int row = blockIdx.x;
    const int tid = threadIdx.x;
    if (row >= rows) return;
    __shared__ float red[256];

    const size_t base = (size_t)row * DMODEL;
    float v0 = x[base + tid]       + (r ? r[base + tid]       : 0.0f);
    float v1 = x[base + tid + 256] + (r ? r[base + tid + 256] : 0.0f);

    red[tid] = v0 + v1;
    __syncthreads();
    for (int s = 128; s > 0; s >>= 1) { if (tid < s) red[tid] += red[tid + s]; __syncthreads(); }
    const float mean = red[0] * (1.0f / DMODEL);
    __syncthreads();

    const float d0 = v0 - mean, d1 = v1 - mean;
    red[tid] = d0 * d0 + d1 * d1;
    __syncthreads();
    for (int s = 128; s > 0; s >>= 1) { if (tid < s) red[tid] += red[tid + s]; __syncthreads(); }
    const float inv = rsqrtf(red[0] * (1.0f / DMODEL) + 1e-5f);
    __syncthreads();

    out[base + tid]       = d0 * inv * g[tid]       + b[tid];
    out[base + tid + 256] = d1 * inv * g[tid + 256] + b[tid + 256];
}

// obj[0][b][:] = emb_table[b][:]  (first 64 rows of the table)
__global__ void gather_emb_kernel(const float* __restrict__ emb, float* __restrict__ obj0)
{
    int i = blockIdx.x * blockDim.x + threadIdx.x;
    if (i < BATCH * DMODEL) obj0[i] = emb[i];
}

// d_out = [box (b-major [B,L,LAT])] ++ [comp (b-major [B,L,1])]
// from workspace buffers laid out token-major [l*B + b].
__global__ void finalize_kernel(const float* __restrict__ boxt,
                                const float* __restrict__ compt,
                                float* __restrict__ out)
{
    int i = blockIdx.x * blockDim.x + threadIdx.x;
    if (i < LLANE * BATCH * LATENT) {
        int j = i & (LATENT - 1);
        int b = (i >> 6) & (BATCH - 1);
        int l = i >> 12;
        out[((size_t)b * LLANE + l) * LATENT + j] = boxt[i];
    }
    if (i < LLANE * BATCH) {
        int b = i & (BATCH - 1);
        int l = i >> 6;
        out[(size_t)LLANE * BATCH * LATENT + (size_t)b * LLANE + l] = compt[i];
    }
}

// ---------------------------------------------------------------------------
// Host orchestration
// ---------------------------------------------------------------------------
static inline void gemm(hipStream_t st, const float* A, const float* W, const float* b,
                        float* C, int M, int N, int K, int act)
{
    dim3 grid((N + 63) / 64, M / 64);   // M is always a multiple of 64 here
    gemm_wmma_kernel<<<grid, 128, 0, st>>>(A, W, b, C, M, N, K, act);
}

extern "C" void kernel_launch(void* const* d_in, const int* in_sizes, int n_in,
                              void* d_out, int out_size, void* d_ws, size_t ws_size,
                              hipStream_t stream)
{
    (void)in_sizes; (void)n_in; (void)out_size; (void)ws_size;
    auto F = [&](int i) { return (const float*)d_in[i]; };

    // Input index map (setup_inputs dict order, nested dicts flattened in order)
    const float* latent = F(0);          // [64, 64000]
    const float* emb    = F(1);
    const float* fW1 = F(2), *fb1 = F(3), *fW2 = F(4), *fb2 = F(5);
    // encoder
    const float* eWq = F(6),  *ebq = F(7),  *eWk = F(8),  *ebk = F(9);
    const float* eWv = F(10), *ebv = F(11), *eWo = F(12), *ebo = F(13);
    const float* eW1 = F(14), *ebf1 = F(15), *eW2 = F(16), *ebf2 = F(17);
    const float* eg1 = F(18), *eb1 = F(19), *eg2 = F(20), *eb2 = F(21);
    const float* egf = F(22), *ebf = F(23);
    // decoder self / cross attention
    const float* sWq = F(24), *sbq = F(25), *sWk = F(26), *sbk = F(27);
    const float* sWv = F(28), *sbv = F(29), *sWo = F(30), *sbo = F(31);
    const float* cWq = F(32), *cbq = F(33), *cWk = F(34), *cbk = F(35);
    const float* cWv = F(36), *cbv = F(37), *cWo = F(38), *cbo = F(39);
    const float* dW1 = F(40), *dbf1 = F(41), *dW2 = F(42), *dbf2 = F(43);
    const float* dg1 = F(44), *db1 = F(45), *dg2 = F(46), *db2 = F(47);
    const float* dg3 = F(48), *db3 = F(49), *dgf = F(50), *dbf = F(51);
    const float* stW1 = F(52), *stb1 = F(53), *stW2 = F(54), *stb2 = F(55);
    const float* bxW1 = F(56), *bxb1 = F(57), *bxW2 = F(58), *bxb2 = F(59);
    const float* cpW1 = F(60), *cpb1 = F(61), *cpW2 = F(62), *cpb2 = F(63);

    // Workspace carve-up (floats)
    float* ws = (float*)d_ws;
    size_t off = 0;
    auto alloc = [&](size_t nf) { float* p = ws + off; off += nf; return p; };
    const size_t ROWS = (size_t)LLANE * BATCH * DMODEL;   // 768 x 512
    float* h1   = alloc((size_t)BATCH * 1024);
    float* zlat = alloc((size_t)BATCH * DMODEL);
    float* obj  = alloc((size_t)(LLANE + 1) * BATCH * DMODEL);  // [13][64][512]
    float* memb = alloc(ROWS);
    float* tQ   = alloc(ROWS);
    float* tK   = alloc(ROWS);
    float* tV   = alloc(ROWS);
    float* tA   = alloc(ROWS);
    float* tP   = alloc(ROWS);
    float* tX1  = alloc(ROWS);
    float* tH   = alloc(ROWS);
    float* tY   = alloc(ROWS);                             // decoder scratch "dx"
    float* boxt = alloc((size_t)LLANE * BATCH * LATENT);
    float* cmpt = alloc((size_t)LLANE * BATCH);

    const int FLAT_IN = 512 * 25 * 5;                      // 64000

    // 1) Flatten MLP -> latent_z (HBM-bound: W1 streamed exactly once, M=64)
    gemm(stream, latent, fW1, fb1, h1, BATCH, 1024, FLAT_IN, 1);
    gemm(stream, h1, fW2, fb2, zlat, BATCH, DMODEL, 1024, 0);

    // 2) obj[0] = emb_table[b]
    gather_emb_kernel<<<(BATCH * DMODEL + 255) / 256, 256, 0, stream>>>(emb, obj);

    // 3) Autoregressive loop
    for (int idx = 0; idx < LLANE; ++idx) {
        const int s  = idx + 1;
        const int Ms = s * BATCH;          // encoder token rows (multiple of 64)
        const float* X = obj;              // rows [0, Ms)

        // ---- encoder (post-norm) ----
        gemm(stream, X, eWq, ebq, tQ, Ms, DMODEL, DMODEL, 0);
        gemm(stream, X, eWk, ebk, tK, Ms, DMODEL, DMODEL, 0);
        gemm(stream, X, eWv, ebv, tV, Ms, DMODEL, DMODEL, 0);
        attn_kernel<<<s * BATCH * NHEAD, 64, 0, stream>>>(tQ, tK, tV, tA, s, s);
        gemm(stream, tA, eWo, ebo, tP, Ms, DMODEL, DMODEL, 0);
        add_ln_kernel<<<Ms, 256, 0, stream>>>(tX1, X, tP, eg1, eb1, Ms);
        gemm(stream, tX1, eW1, ebf1, tH, Ms, DMODEL, DMODEL, 2);    // gelu
        gemm(stream, tH, eW2, ebf2, tP, Ms, DMODEL, DMODEL, 0);
        add_ln_kernel<<<Ms, 256, 0, stream>>>(tX1, tX1, tP, eg2, eb2, Ms);
        add_ln_kernel<<<Ms, 256, 0, stream>>>(memb, tX1, nullptr, egf, ebf, Ms);

        // ---- decoder (query = latent_z, len 1) ----
        float* dx = tY;
        gemm(stream, zlat, sWq, sbq, tQ, BATCH, DMODEL, DMODEL, 0);
        gemm(stream, zlat, sWk, sbk, tK, BATCH, DMODEL, DMODEL, 0);
        gemm(stream, zlat, sWv, sbv, tV, BATCH, DMODEL, DMODEL, 0);
        attn_kernel<<<BATCH * NHEAD, 64, 0, stream>>>(tQ, tK, tV, tA, 1, 1);
        gemm(stream, tA, sWo, sbo, tP, BATCH, DMODEL, DMODEL, 0);
        add_ln_kernel<<<BATCH, 256, 0, stream>>>(dx, zlat, tP, dg1, db1, BATCH);

        gemm(stream, dx, cWq, cbq, tQ, BATCH, DMODEL, DMODEL, 0);
        gemm(stream, memb, cWk, cbk, tK, Ms, DMODEL, DMODEL, 0);
        gemm(stream, memb, cWv, cbv, tV, Ms, DMODEL, DMODEL, 0);
        attn_kernel<<<BATCH * NHEAD, 64, 0, stream>>>(tQ, tK, tV, tA, 1, s);
        gemm(stream, tA, cWo, cbo, tP, BATCH, DMODEL, DMODEL, 0);
        add_ln_kernel<<<BATCH, 256, 0, stream>>>(dx, dx, tP, dg2, db2, BATCH);

        gemm(stream, dx, dW1, dbf1, tH, BATCH, DMODEL, DMODEL, 2);  // gelu
        gemm(stream, tH, dW2, dbf2, tP, BATCH, DMODEL, DMODEL, 0);
        add_ln_kernel<<<BATCH, 256, 0, stream>>>(dx, dx, tP, dg3, db3, BATCH);
        add_ln_kernel<<<BATCH, 256, 0, stream>>>(dx, dx, nullptr, dgf, dbf, BATCH);

        // ---- per-step MLP -> obj[idx+1] ----
        const float* W1i = stW1 + (size_t)idx * DMODEL * DMODEL;
        const float* b1i = stb1 + (size_t)idx * DMODEL;
        const float* W2i = stW2 + (size_t)idx * DMODEL * DMODEL;
        const float* b2i = stb2 + (size_t)idx * DMODEL;
        gemm(stream, dx, W1i, b1i, tH, BATCH, DMODEL, DMODEL, 1);   // relu
        gemm(stream, tH, W2i, b2i, obj + (size_t)(idx + 1) * BATCH * DMODEL,
             BATCH, DMODEL, DMODEL, 1);                             // relu
    }

    // 4) Heads on feats = obj[1..12] (contiguous rows [B, 13B))
    const float* feats = obj + (size_t)BATCH * DMODEL;
    const int MF = LLANE * BATCH;                                   // 768
    gemm(stream, feats, bxW1, bxb1, tH, MF, DMODEL, DMODEL, 1);
    gemm(stream, tH, bxW2, bxb2, boxt, MF, LATENT, DMODEL, 0);
    gemm(stream, feats, cpW1, cpb1, tQ, MF, 128, DMODEL, 1);
    gemm(stream, tQ, cpW2, cpb2, cmpt, MF, 1, 128, 0);

    // 5) Transpose token-major -> batch-major into d_out
    finalize_kernel<<<(LLANE * BATCH * LATENT + 255) / 256, 256, 0, stream>>>(
        boxt, cmpt, (float*)d_out);
}
